// MultiHeadLatentAttention_45526653337965
// MI455X (gfx1250) — matrix operand order repaired and checked by reading, not compiled, not run on hardware
//
#include <hip/hip_runtime.h>
#include <stdint.h>

#define USE_TDM 1

// ---------------- problem constants ----------------
#define CB   2
#define CS   2048
#define CHID 2048
#define CH   16
#define CKVH 4
#define CDH  128
#define CL   64

typedef __attribute__((ext_vector_type(16))) __bf16 v16bf;
typedef __attribute__((ext_vector_type(8)))  float  v8f;
typedef __attribute__((ext_vector_type(4)))  uint32_t u32x4;
typedef __attribute__((ext_vector_type(8)))  uint32_t u32x8;

union Frag { uint32_t u[8]; v16bf v; };

__device__ __forceinline__ uint16_t f2bf(float f) {
  uint32_t u = __float_as_uint(f);
  u = (u + 0x7FFFu + ((u >> 16) & 1u)) >> 16;   // round-to-nearest-even
  return (uint16_t)u;
}
__device__ __forceinline__ float bf2f(uint16_t h) {
  return __uint_as_float(((uint32_t)h) << 16);
}
// 16-bit A-matrix 16x32 layout: lane half selects K+8; vgpr 0..3 -> K 0..7, vgpr 4..7 -> K 16..23
__device__ __forceinline__ int a_kpair(int v, int hf) {
  return (v < 4 ? 2 * v : 16 + 2 * (v - 4)) + 8 * hf;
}
// 16-bit B-matrix 32x16 layout (matches A): lane half selects K+8; vgpr 0..3 -> K 0..7, vgpr 4..7 -> K 16..23
__device__ __forceinline__ int b_kpair(int v, int hf) {
  return (v < 4 ? 2 * v : 16 + 2 * (v - 4)) + 8 * hf;
}
__device__ __forceinline__ v8f wmma_bf16(const Frag& a, const Frag& b, v8f c) {
  return __builtin_amdgcn_wmma_f32_16x16x32_bf16(
      false, a.v, false, b.v, (short)0, c, false, false);
}

#if USE_TDM
// ---- Tensor Data Mover: 2D bf16 tile global -> LDS (issued once per wave) ----
// D# group0: [1:0]=count(1) | [63:32]=lds_addr | [120:64]=global_addr | [127:126]=type(2)
// D# group1: [17:16]=data_size(1: 2B) | [79:48]=tensor_dim0 | [111:80]=tensor_dim1
//            | [127:112]=tile_dim0 | [143:128]=tile_dim1 | [207:160]=tensor_dim0_stride
__device__ __forceinline__ void tdm_load_2d_bf16(const uint16_t* gsrc,
                                                 uint32_t ldsOff,
                                                 uint32_t tileW, uint32_t tileH,
                                                 uint32_t rowStrideElems) {
  const uint64_t ga = (uint64_t)(size_t)gsrc;
  u32x4 g0;
  g0[0] = 1u;                                           // count=1 (valid)
  g0[1] = ldsOff;                                       // LDS byte address
  g0[2] = (uint32_t)ga;                                 // global addr [31:0]
  g0[3] = (uint32_t)((ga >> 32) & 0x1FFFFFFu) | 0x80000000u;  // addr[56:32] | type=2
  u32x8 g1;
  g1[0] = 0x00010000u;                                  // data_size = 1 -> 2 bytes
  g1[1] = (tileW & 0xFFFFu) << 16;                      // tensor_dim0[15:0]
  g1[2] = (tileW >> 16) | ((tileH & 0xFFFFu) << 16);    // tensor_dim0[31:16] | tensor_dim1[15:0]
  g1[3] = (tileH >> 16) | (tileW << 16);                // tensor_dim1[31:16] | tile_dim0
  g1[4] = tileH & 0xFFFFu;                              // tile_dim1 (tile_dim2 = 0)
  g1[5] = rowStrideElems;                               // tensor_dim0_stride[31:0]
  g1[6] = 0u;                                           // stride0[47:32]=0 | stride1[15:0]=0
  g1[7] = 0u;
  const u32x4 z = {0u, 0u, 0u, 0u};                     // groups 2/3: dims unused
  asm volatile("tensor_load_to_lds %0, %1, %2, %3"
               :: "s"(g0), "s"(g1), "s"(z), "s"(z)
               : "memory");
}
__device__ __forceinline__ void tdm_wait() {
  __builtin_amdgcn_s_wait_tensorcnt(0);
}
#endif

// ---------------- fp32 -> bf16 conversion ----------------
__global__ void cvt_bf16_kernel(const float* __restrict__ src,
                                uint16_t* __restrict__ dst, long n) {
  long i = (long)blockIdx.x * blockDim.x + threadIdx.x;
  if (i < n) dst[i] = f2bf(src[i]);
}

// ---------------- generic bf16 WMMA GEMM ----------------
// C[M,N] = A[M,K] @ B[K,N]; A,B bf16 row-major. 128x128 tile, BK=32, 256 thr (8 waves).
// F32OUT: write f32 and add bias[(row/rowsPerBatch)*N + col].
template <bool F32OUT>
__global__ __launch_bounds__(256) void gemm_bf16_kernel(
    const uint16_t* __restrict__ A, const uint16_t* __restrict__ Bm,
    void* __restrict__ Cv, int M, int N, int K,
    const float* __restrict__ bias, int rowsPerBatch) {
  __shared__ uint16_t lds_a[128][32];    // A tile, row-major
  __shared__ uint16_t lds_bT[128][32];   // B tile, transposed [n][k]
  const int tid = threadIdx.x;
  const int lane = tid & 31, wave = tid >> 5;
  const int hf = lane >> 4, ln = lane & 15;
  const int wr = wave & 3, wc = wave >> 2;       // wave tile: rows wr*32, cols wc*64
  const int bm0 = blockIdx.x * 128;
  const int bn0 = blockIdx.y * 128;

  v8f acc[2][4] = {};

  const int br = tid >> 3, bc = (tid & 7) << 4;  // B staging: 32 rows x 8 segs
#if USE_TDM
  const uint32_t ldsAoff = (uint32_t)(size_t)&lds_a[0][0];
#else
  const int ar = tid >> 1, ac = (tid & 1) << 4;  // A staging: 128 rows x 2 segs
#endif

  for (int kb = 0; kb < K; kb += 32) {
#if USE_TDM
    // stage A tile 128x32 via Tensor Data Mover (wave 0 issues the DMA)
    if (tid < 32)
      tdm_load_2d_bf16(A + (size_t)bm0 * K + kb, ldsAoff, 32u, 128u, (uint32_t)K);
#else
    {  // stage A tile 128x32 manually
      const uint16_t* src = A + (size_t)(bm0 + ar) * K + kb + ac;
      uint4 d0 = *(const uint4*)(src);
      uint4 d1 = *(const uint4*)(src + 8);
      *(uint4*)&lds_a[ar][ac]     = d0;
      *(uint4*)&lds_a[ar][ac + 8] = d1;
    }
#endif
    {  // stage B tile 32x128, transposed into lds_bT (TDM cannot transpose)
      const uint16_t* src = Bm + (size_t)(kb + br) * N + bn0 + bc;
      uint4 d0 = *(const uint4*)(src);
      uint4 d1 = *(const uint4*)(src + 8);
      uint16_t tmp[16];
      *(uint4*)&tmp[0] = d0;
      *(uint4*)&tmp[8] = d1;
#pragma unroll
      for (int i = 0; i < 16; ++i) lds_bT[bc + i][br] = tmp[i];
    }
    if (kb + 32 < K)  // hint next B tile into cache (global_prefetch_b8)
      __builtin_prefetch(Bm + (size_t)(kb + 32 + br) * N + bn0 + bc, 0, 0);
#if USE_TDM
    if (tid < 32) tdm_wait();  // TENSORcnt -> 0 in the issuing wave
#endif
    __syncthreads();

    Frag af[2], bf[4];
#pragma unroll
    for (int i = 0; i < 2; ++i) {
      const int m = wr * 32 + i * 16 + ln;
#pragma unroll
      for (int v = 0; v < 8; ++v)
        af[i].u[v] = *(const uint32_t*)&lds_a[m][a_kpair(v, hf)];
    }
#pragma unroll
    for (int j = 0; j < 4; ++j) {
      const int n = wc * 64 + j * 16 + ln;
#pragma unroll
      for (int v = 0; v < 8; ++v)
        bf[j].u[v] = *(const uint32_t*)&lds_bT[n][b_kpair(v, hf)];
    }
#pragma unroll
    for (int i = 0; i < 2; ++i)
#pragma unroll
      for (int j = 0; j < 4; ++j)
        acc[i][j] = wmma_bf16(af[i], bf[j], acc[i][j]);
    __syncthreads();
  }

  // C layout: vgpr r <-> row r + 8*half; lane ln <-> col
#pragma unroll
  for (int i = 0; i < 2; ++i) {
#pragma unroll
    for (int r = 0; r < 8; ++r) {
      const int row = bm0 + wr * 32 + i * 16 + r + 8 * hf;
#pragma unroll
      for (int j = 0; j < 4; ++j) {
        const int col = bn0 + wc * 64 + j * 16 + ln;
        if constexpr (F32OUT) {
          float* C = (float*)Cv;
          C[(size_t)row * N + col] =
              acc[i][j][r] + bias[(size_t)(row / rowsPerBatch) * N + col];
        } else {
          uint16_t* C = (uint16_t*)Cv;
          C[(size_t)row * N + col] = f2bf(acc[i][j][r]);
        }
      }
    }
  }
}

// ---------------- RoPE + reshape to head-major ----------------
// in:  proj [B*S, NH*128] bf16 ; out: [B, NH, S, 128] bf16 (rotated)
__global__ void rope_reshape_kernel(const uint16_t* __restrict__ proj,
                                    uint16_t* __restrict__ out, int NH) {
  size_t idx = (size_t)blockIdx.x * blockDim.x + threadIdx.x;
  const int d = (int)(idx & 127);
  size_t t = idx >> 7;
  const int s = (int)(t & (CS - 1)); t >>= 11;
  const int nh = (int)(t % NH);
  const int b = (int)(t / NH);
  const uint16_t* row = proj + ((size_t)b * CS + s) * ((size_t)NH * CDH) + (size_t)nh * CDH;
  const float x = bf2f(row[d]);
  const float xr = (d < 64) ? -bf2f(row[d + 64]) : bf2f(row[d - 64]);
  const int j = d & 63;
  // inv_freq = theta^(-j/64); log(10000)/64 = 0.14391156831...
  const float angle = (float)s * __expf(-(float)j * 0.14391156831212787f);
  float sn, cs;
  __sincosf(angle, &sn, &cs);
  out[idx] = f2bf(x * cs + xr * sn);
}

// ---------------- flash attention (bf16 WMMA, online softmax) ----------------
// q: [B,H,S,128] roped; k: [B,KVH,S,128] roped; v: raw proj [B*S, KVH*128]
// out: attn [B*S, H*128] bf16
__global__ __launch_bounds__(128) void attn_kernel(
    const uint16_t* __restrict__ qr, const uint16_t* __restrict__ kr,
    const uint16_t* __restrict__ vproj, uint16_t* __restrict__ attn) {
  __shared__ uint16_t lds_k[32][128];     // 32 keys x 128 dh
  __shared__ uint16_t lds_vT[128][32];    // transposed: [dh][key]
  __shared__ uint16_t lds_p[4][16][32];   // per-wave P repack scratch
  const int tid = threadIdx.x, lane = tid & 31, wave = tid >> 5;
  const int hf = lane >> 4, ln = lane & 15;
  const int b = blockIdx.z, h = blockIdx.y;
  const int kvh = h >> 2;                 // H/KVH = 4
  const int q0 = blockIdx.x * 64 + wave * 16;

  // load Q fragments for 16 rows x 128 dh (4 k-chunks of 32)
  Frag qf[4];
  const uint16_t* qbase = qr + (((size_t)b * CH + h) * CS + q0 + ln) * CDH;
#pragma unroll
  for (int kc = 0; kc < 4; ++kc)
#pragma unroll
    for (int v = 0; v < 8; ++v)
      qf[kc].u[v] = *(const uint32_t*)(qbase + kc * 32 + a_kpair(v, hf));

  float mrow[8], lrow[8];
  v8f o[8] = {};
#pragma unroll
  for (int r = 0; r < 8; ++r) { mrow[r] = -1e30f; lrow[r] = 0.0f; }

  const float scale = 0.08838834764831845f;  // 1/sqrt(128)
  const int sr = tid >> 2, scs = (tid & 3) << 5;  // staging: 32 rows x 4 segs of 32
#if USE_TDM
  const uint32_t ldsKoff = (uint32_t)(size_t)&lds_k[0][0];
  const uint16_t* kbase = kr + (((size_t)b * CKVH + kvh) * CS) * CDH;
#endif

  for (int kb = 0; kb < CS; kb += 32) {
#if USE_TDM
    // K tile is 32x128 contiguous bf16 -> one 4096-element 1D TDM transfer
    if (tid < 32) tdm_load_2d_bf16(kbase + (size_t)kb * CDH, ldsKoff, 4096u, 1u, 4096u);
#else
    {  // stage K tile
      const uint16_t* src = kr + (((size_t)b * CKVH + kvh) * CS + kb + sr) * CDH + scs;
#pragma unroll
      for (int q = 0; q < 4; ++q)
        *(uint4*)&lds_k[sr][scs + q * 8] = *(const uint4*)(src + q * 8);
    }
#endif
    {  // stage V tile transposed (TDM cannot transpose)
      const uint16_t* src =
          vproj + ((size_t)b * CS + kb + sr) * (CKVH * CDH) + kvh * CDH + scs;
      uint16_t tmp[32];
#pragma unroll
      for (int q = 0; q < 4; ++q) *(uint4*)&tmp[q * 8] = *(const uint4*)(src + q * 8);
#pragma unroll
      for (int i = 0; i < 32; ++i) lds_vT[scs + i][sr] = tmp[i];
    }
#if USE_TDM
    if (tid < 32) tdm_wait();
#endif
    __syncthreads();

    // scores: S = Q (16x128) @ K^T (128x32) -> two 16x16 key groups
    v8f s0 = {}, s1 = {};
#pragma unroll
    for (int kc = 0; kc < 4; ++kc) {
      Frag b0, b1;
#pragma unroll
      for (int v = 0; v < 8; ++v) {
        const int kp = kc * 32 + b_kpair(v, hf);
        b0.u[v] = *(const uint32_t*)&lds_k[ln][kp];
        b1.u[v] = *(const uint32_t*)&lds_k[ln + 16][kp];
      }
      s0 = wmma_bf16(qf[kc], b0, s0);
      s1 = wmma_bf16(qf[kc], b1, s1);
    }

    // online softmax; row r+8*hf stats reduced across the 16-lane half
#pragma unroll
    for (int r = 0; r < 8; ++r) {
      const float a0 = s0[r] * scale, a1 = s1[r] * scale;
      float mx = fmaxf(a0, a1);
#pragma unroll
      for (int d = 1; d < 16; d <<= 1) mx = fmaxf(mx, __shfl_xor(mx, d, 32));
      const float mnew = fmaxf(mrow[r], mx);
      const float alpha = __expf(mrow[r] - mnew);
      const float p0 = __expf(a0 - mnew);
      const float p1 = __expf(a1 - mnew);
      float psum = p0 + p1;
#pragma unroll
      for (int d = 1; d < 16; d <<= 1) psum += __shfl_xor(psum, d, 32);
      lrow[r] = lrow[r] * alpha + psum;
      mrow[r] = mnew;
#pragma unroll
      for (int cg = 0; cg < 8; ++cg) o[cg][r] = o[cg][r] * alpha;
      lds_p[wave][r + 8 * hf][ln]      = f2bf(p0);
      lds_p[wave][r + 8 * hf][ln + 16] = f2bf(p1);
    }

    // P (16x32, A layout via per-wave LDS) @ V (32x128)
    Frag pf;
#pragma unroll
    for (int v = 0; v < 8; ++v)
      pf.u[v] = *(const uint32_t*)&lds_p[wave][ln][a_kpair(v, hf)];
#pragma unroll
    for (int cg = 0; cg < 8; ++cg) {
      Frag vf;
#pragma unroll
      for (int v = 0; v < 8; ++v)
        vf.u[v] = *(const uint32_t*)&lds_vT[cg * 16 + ln][b_kpair(v, hf)];
      o[cg] = wmma_bf16(pf, vf, o[cg]);
    }
    __syncthreads();
  }

  // normalize and store: attn[b*S+row][h*128 + col]
#pragma unroll
  for (int r = 0; r < 8; ++r) {
    const float inv = 1.0f / lrow[r];
    const int row = q0 + r + 8 * hf;
    uint16_t* dst = attn + ((size_t)b * CS + row) * CHID + h * CDH;
#pragma unroll
    for (int cg = 0; cg < 8; ++cg)
      dst[cg * 16 + ln] = f2bf(o[cg][r] * inv);
  }
}

// ---------------- latent branch (tiny, f32 VALU) ----------------
// lat[256,128] = latent[256,128] @ w_latent[128,128]
__global__ __launch_bounds__(128) void latent_proj_kernel(
    const float* __restrict__ latent, const float* __restrict__ wlat,
    float* __restrict__ lat) {
  const int row = blockIdx.x, col = threadIdx.x;
  __shared__ float lrow[128];
  lrow[col] = latent[(size_t)row * 128 + col];
  __syncthreads();
  float acc = 0.0f;
  for (int k = 0; k < 128; ++k) acc += lrow[k] * wlat[(size_t)k * 128 + col];
  lat[(size_t)row * 128 + col] = acc;
}

// latent cross-attn: one block per (b,kvh,l); softmax over S keys
__global__ __launch_bounds__(128) void latent_attn_kernel(
    const float* __restrict__ lat, const uint16_t* __restrict__ kr,
    const uint16_t* __restrict__ vproj, float* __restrict__ latout) {
  const int b = blockIdx.x >> 8;
  const int kvh = (blockIdx.x >> 6) & 3;
  const int l = blockIdx.x & 63;
  const int tid = threadIdx.x;
  __shared__ float q[128];
  __shared__ float probs[CS];
  __shared__ float red[128];
  q[tid] = lat[((size_t)kvh * CL + l) * CDH + tid];
  __syncthreads();
  const uint16_t* kbase = kr + (((size_t)b * CKVH + kvh) * CS) * CDH;
  float lmax = -1e30f;
  for (int s = tid; s < CS; s += 128) {
    const uint16_t* krow = kbase + (size_t)s * CDH;
    float acc = 0.0f;
    for (int d = 0; d < CDH; ++d) acc += q[d] * bf2f(krow[d]);
    acc *= 0.08838834764831845f;
    probs[s] = acc;
    lmax = fmaxf(lmax, acc);
  }
  red[tid] = lmax;
  __syncthreads();
  for (int st = 64; st > 0; st >>= 1) {
    if (tid < st) red[tid] = fmaxf(red[tid], red[tid + st]);
    __syncthreads();
  }
  const float gmax = red[0];
  __syncthreads();
  float lsum = 0.0f;
  for (int s = tid; s < CS; s += 128) {
    const float p = __expf(probs[s] - gmax);
    probs[s] = p;
    lsum += p;
  }
  red[tid] = lsum;
  __syncthreads();
  for (int st = 64; st > 0; st >>= 1) {
    if (tid < st) red[tid] += red[tid + st];
    __syncthreads();
  }
  const float inv = 1.0f / red[0];
  __syncthreads();
  const uint16_t* vbase = vproj + (size_t)b * CS * (CKVH * CDH) + kvh * CDH + tid;
  float acc = 0.0f;
  for (int s = 0; s < CS; ++s) acc += probs[s] * bf2f(vbase[(size_t)s * (CKVH * CDH)]);
  latout[((((size_t)b * CKVH + kvh) * CL) + l) * CDH + tid] = acc * inv;
}

// mean over L (folded): latsum[b][c] = (1/L) sum_l latout[b][c>>7][l][c&127]
__global__ void latent_sum_kernel(const float* __restrict__ latout,
                                  float* __restrict__ latsum) {
  const int i = blockIdx.x * blockDim.x + threadIdx.x;
  if (i >= CB * CKVH * CDH) return;
  const int b = i >> 9, c = i & 511, kvh = c >> 7, d = c & 127;
  float acc = 0.0f;
  for (int l = 0; l < CL; ++l)
    acc += latout[((((size_t)b * CKVH + kvh) * CL) + l) * CDH + d];
  latsum[i] = acc * (1.0f / CL);
}

// latvec[b][n] = sum_c latsum[b][c] * wlo[c][n]
__global__ __launch_bounds__(256) void latent_vec_kernel(
    const float* __restrict__ latsum, const float* __restrict__ wlo,
    float* __restrict__ latvec) {
  const int i = blockIdx.x * 256 + threadIdx.x;
  const int b = i >> 11, n = i & (CHID - 1);
  float acc = 0.0f;
  for (int c = 0; c < CKVH * CDH; ++c)
    acc += latsum[(size_t)b * (CKVH * CDH) + c] * wlo[(size_t)c * CHID + n];
  latvec[i] = acc;
}

// ---------------- host orchestration ----------------
extern "C" void kernel_launch(void* const* d_in, const int* in_sizes, int n_in,
                              void* d_out, int out_size, void* d_ws, size_t ws_size,
                              hipStream_t stream) {
  (void)in_sizes; (void)n_in; (void)out_size; (void)ws_size;
  const float* hs     = (const float*)d_in[0];
  const float* latent = (const float*)d_in[1];
  const float* wlat   = (const float*)d_in[2];
  const float* wq     = (const float*)d_in[3];
  const float* wk     = (const float*)d_in[4];
  const float* wv     = (const float*)d_in[5];
  const float* wo     = (const float*)d_in[6];
  const float* wlo    = (const float*)d_in[7];
  float* out = (float*)d_out;

  const size_t SZ_HS = (size_t)CB * CS * CHID;            // 8,388,608
  const size_t SZ_WQ = (size_t)CHID * CH * CDH;           // 4,194,304
  const size_t SZ_WK = (size_t)CHID * CKVH * CDH;         // 1,048,576
  const size_t SZ_WO = (size_t)CHID * CHID;               // 4,194,304
  const size_t SZ_QP = (size_t)CB * CS * CH * CDH;        // 8,388,608
  const size_t SZ_KP = (size_t)CB * CS * CKVH * CDH;      // 2,097,152

  uint8_t* p = (uint8_t*)d_ws;
  uint16_t* hs_bf  = (uint16_t*)p; p += SZ_HS * 2;
  uint16_t* wq_bf  = (uint16_t*)p; p += SZ_WQ * 2;
  uint16_t* wk_bf  = (uint16_t*)p; p += SZ_WK * 2;
  uint16_t* wv_bf  = (uint16_t*)p; p += SZ_WK * 2;
  uint16_t* wo_bf  = (uint16_t*)p; p += SZ_WO * 2;
  uint16_t* qproj  = (uint16_t*)p; p += SZ_QP * 2;
  uint16_t* kproj  = (uint16_t*)p; p += SZ_KP * 2;
  uint16_t* vproj  = (uint16_t*)p; p += SZ_KP * 2;
  uint16_t* qrope  = (uint16_t*)p; p += SZ_QP * 2;
  uint16_t* krope  = (uint16_t*)p; p += SZ_KP * 2;
  float* lat_f32   = (float*)p; p += (size_t)CKVH * CL * CDH * 4;
  float* latout    = (float*)p; p += (size_t)CB * CKVH * CL * CDH * 4;
  float* latsum    = (float*)p; p += (size_t)CB * CKVH * CDH * 4;
  float* latvec    = (float*)p; p += (size_t)CB * CHID * 4;
  uint16_t* attn_bf = qproj;  // alias: q-projection dead after RoPE

  auto cvt = [&](const float* s, uint16_t* d, size_t n) {
    cvt_bf16_kernel<<<dim3((unsigned)((n + 255) / 256)), dim3(256), 0, stream>>>(s, d, (long)n);
  };
  cvt(hs, hs_bf, SZ_HS);
  cvt(wq, wq_bf, SZ_WQ);
  cvt(wk, wk_bf, SZ_WK);
  cvt(wv, wv_bf, SZ_WK);
  cvt(wo, wo_bf, SZ_WO);

  // Q/K/V projections (bf16 WMMA)
  gemm_bf16_kernel<false><<<dim3(32, 16), 256, 0, stream>>>(
      hs_bf, wq_bf, qproj, CB * CS, CH * CDH, CHID, nullptr, 1);
  gemm_bf16_kernel<false><<<dim3(32, 4), 256, 0, stream>>>(
      hs_bf, wk_bf, kproj, CB * CS, CKVH * CDH, CHID, nullptr, 1);
  gemm_bf16_kernel<false><<<dim3(32, 4), 256, 0, stream>>>(
      hs_bf, wv_bf, vproj, CB * CS, CKVH * CDH, CHID, nullptr, 1);

  // RoPE + head-major reshape
  rope_reshape_kernel<<<dim3((unsigned)(SZ_QP / 256)), dim3(256), 0, stream>>>(qproj, qrope, CH);
  rope_reshape_kernel<<<dim3((unsigned)(SZ_KP / 256)), dim3(256), 0, stream>>>(kproj, krope, CKVH);

  // flash attention (writes attn_bf, aliasing qproj)
  attn_kernel<<<dim3(CS / 64, CH, CB), dim3(128), 0, stream>>>(qrope, krope, vproj, attn_bf);

  // latent branch
  latent_proj_kernel<<<dim3(CKVH * CL), dim3(128), 0, stream>>>(latent, wlat, lat_f32);
  latent_attn_kernel<<<dim3(CB * CKVH * CL), dim3(128), 0, stream>>>(lat_f32, krope, vproj, latout);
  latent_sum_kernel<<<dim3(4), dim3(256), 0, stream>>>(latout, latsum);
  latent_vec_kernel<<<dim3(CB * CHID / 256), dim3(256), 0, stream>>>(latsum, wlo, latvec);

  // output projection + broadcast latent add (f32 out)
  gemm_bf16_kernel<true><<<dim3(32, 16), 256, 0, stream>>>(
      attn_bf, wo_bf, out, CB * CS, CHID, CHID, latvec, CS);
}
